// MultiHeadAttention_2791728742486
// MI455X (gfx1250) — matrix operand order repaired and checked by reading, not compiled, not run on hardware
//
#include <hip/hip_runtime.h>

typedef _Float16 h16_t;
typedef __attribute__((ext_vector_type(16))) _Float16 v16h;
typedef __attribute__((ext_vector_type(8)))  _Float16 v8h;
typedef __attribute__((ext_vector_type(8)))  float    v8f;

#define DIM   1024
#define SEQ   2048
#define BATCH 4
#define HEADS 16
#define DHEAD 64

// ---- WMMA helpers -----------------------------------------------------------

__device__ __forceinline__ v8f wmma_f16(v16h a, v16h b, v8f c) {
  // D = A(16x32 f16) x B(32x16 f16) + C(16x16 f32)
  return __builtin_amdgcn_wmma_f32_16x16x32_f16(
      /*neg_a=*/false, a, /*neg_b=*/false, b,
      /*c_mod=*/(short)0, c, /*reuse_a=*/false, /*reuse_b=*/false);
}

// A fragment in this code's k order (matches B; not the ISA table order): lane holds row M=lane&15;
// lane<16: K = 0..15 contiguous; lane>=16: K = 16..31 contiguous.
// Caller passes pointer already offset by (lane>>4)*16 halves within the K-chunk.
__device__ __forceinline__ v16h load_fragA(const h16_t* p) {
  v8h lo = *(const v8h*)(p);
  v8h hi = *(const v8h*)(p + 8);
  return __builtin_shufflevector(lo, hi, 0,1,2,3,4,5,6,7,8,9,10,11,12,13,14,15);
}

// B fragment in this code's k order (matches A; not the ISA table order): lane holds col N=lane&15;
// lane<16: K = 0..15 contiguous; lane>=16: K = 16..31 contiguous.
// Caller passes pointer already offset by (lane>>4)*16 halves within the K-chunk.
__device__ __forceinline__ v16h load_fragB(const h16_t* p) {
  v8h lo = *(const v8h*)(p);
  v8h hi = *(const v8h*)(p + 8);
  return __builtin_shufflevector(lo, hi, 0,1,2,3,4,5,6,7,8,9,10,11,12,13,14,15);
}

// ---- Kernel 1: LayerNorm -> f16 --------------------------------------------

__global__ void __launch_bounds__(256)
ln_kernel(const float* __restrict__ x, const float* __restrict__ w,
          const float* __restrict__ bias, h16_t* __restrict__ h) {
  const int row = blockIdx.x;          // 0..8191 (b*2048+n)
  const int t   = threadIdx.x;         // 256 threads, 4 elems each
  const float* xr = x + (size_t)row * DIM;
  float v0[4];
  float s = 0.f;
#pragma unroll
  for (int i = 0; i < 4; ++i) { v0[i] = xr[t + 256 * i]; s += v0[i]; }
  __shared__ float red[256];
  red[t] = s; __syncthreads();
  for (int off = 128; off > 0; off >>= 1) {
    if (t < off) red[t] += red[t + off];
    __syncthreads();
  }
  const float mu = red[0] * (1.0f / DIM);
  __syncthreads();
  float sq = 0.f;
#pragma unroll
  for (int i = 0; i < 4; ++i) { float d = v0[i] - mu; sq += d * d; }
  red[t] = sq; __syncthreads();
  for (int off = 128; off > 0; off >>= 1) {
    if (t < off) red[t] += red[t + off];
    __syncthreads();
  }
  const float rstd = rsqrtf(red[0] * (1.0f / DIM) + 1e-5f);
  h16_t* hr = h + (size_t)row * DIM;
#pragma unroll
  for (int i = 0; i < 4; ++i) {
    const int c = t + 256 * i;
    hr[c] = (h16_t)((v0[i] - mu) * rstd * w[c] + bias[c]);
  }
}

// ---- Kernel 2: weight transpose + f16 convert: Wt[n][k] = W[k][n] ----------

__global__ void __launch_bounds__(256)
wt_kernel(const float* __restrict__ W, h16_t* __restrict__ Wt) {
  const int idx = blockIdx.x * 256 + threadIdx.x;  // 0..1048575
  const int n = idx & (DIM - 1), k = idx >> 10;    // read coalesced over n
  Wt[(size_t)n * DIM + k] = (h16_t)W[idx];
}

// ---- Kernel 3: GEMM  Out = h @ W + b  (f16 WMMA, f32 accum) ----------------
// WG = 128 threads = 4 waves (2x2); each wave computes a 64x64 tile (4x4 WMMA
// tiles, 16 WMMAs per 16 b128 loads per K-step) -> WG tile 128x128.
// V_TRANSPOSED (compile-time): store V per-head transposed Vt[(b*H+h)*64+d][seq].

template <int V_TRANSPOSED>
__global__ void __launch_bounds__(128)
gemm_qkv(const h16_t* __restrict__ A, const h16_t* __restrict__ Wt,
         const float* __restrict__ bias, h16_t* __restrict__ Out) {
  const int lane = threadIdx.x & 31;
  const int w    = threadIdx.x >> 5;
  const int lr   = lane & 15;
  const int hiL  = lane >> 4;
  const int m0 = blockIdx.x * 128 + (w >> 1) * 64;
  const int n0 = blockIdx.y * 128 + (w & 1) * 64;

  v8f c[4][4] = {};
  for (int kc = 0; kc < DIM; kc += 32) {
    v16h a[4], b[4];
#pragma unroll
    for (int mt = 0; mt < 4; ++mt)
      a[mt] = load_fragA(A + (size_t)(m0 + mt * 16 + lr) * DIM + kc + hiL * 16);
#pragma unroll
    for (int nt = 0; nt < 4; ++nt)
      b[nt] = load_fragB(Wt + (size_t)(n0 + nt * 16 + lr) * DIM + kc + hiL * 16);
#pragma unroll
    for (int mt = 0; mt < 4; ++mt)
#pragma unroll
      for (int nt = 0; nt < 4; ++nt)
        c[mt][nt] = wmma_f16(a[mt], b[nt], c[mt][nt]);
  }
#pragma unroll
  for (int mt = 0; mt < 4; ++mt)
#pragma unroll
    for (int nt = 0; nt < 4; ++nt) {
      const int n = n0 + nt * 16 + lr;
      const float bn = bias[n];
#pragma unroll
      for (int r = 0; r < 8; ++r) {
        const int m = m0 + mt * 16 + r + hiL * 8;
        const float val = c[mt][nt][r] + bn;
        if (V_TRANSPOSED) {
          const int bb = m >> 11, tok = m & (SEQ - 1);
          const int hh = n >> 6, d = n & (DHEAD - 1);
          Out[((size_t)((bb * HEADS + hh) * DHEAD + d)) * SEQ + tok] = (h16_t)val;
        } else {
          Out[(size_t)m * DIM + n] = (h16_t)val;
        }
      }
    }
}

// ---- Kernel 4: flash attention ---------------------------------------------
// grid = (SEQ/128, BATCH*HEADS); WG = 4 waves; each wave owns 32 query rows,
// streams 64-key blocks: S = (Q/8)K^T (16 WMMAs), online softmax,
// P -> LDS -> A-frags, O += P V (16 WMMAs) with V stored per-head transposed.

__global__ void __launch_bounds__(128)
attn_kernel(const h16_t* __restrict__ Q, const h16_t* __restrict__ K,
            const h16_t* __restrict__ Vt, float* __restrict__ Out) {
  const int lane = threadIdx.x & 31;
  const int w    = threadIdx.x >> 5;
  const int lr   = lane & 15;
  const int hiL  = lane >> 4;
  const int bh = blockIdx.y;            // 0..63
  const int b  = bh >> 4, h = bh & 15;
  const int q0 = blockIdx.x * 128 + w * 32;

  // per-wave P staging: 32x64 f16, row stride 72 halves (144 B, 16B-aligned)
  __shared__ __align__(16) h16_t pbuf[4][32 * 72];
  h16_t* P = pbuf[w];

  // Q fragments, pre-scaled by 1/sqrt(64) = 0.125 (exact in f16)
  v16h aq[2][2];
#pragma unroll
  for (int mt = 0; mt < 2; ++mt)
#pragma unroll
    for (int kc = 0; kc < 2; ++kc) {
      v16h f = load_fragA(Q + (size_t)(b * SEQ + q0 + mt * 16 + lr) * DIM +
                          h * DHEAD + kc * 32 + hiL * 16);
#pragma unroll
      for (int i = 0; i < 16; ++i) f[i] = f[i] * (_Float16)0.125;
      aq[mt][kc] = f;
    }

  float mrun[2][8], lrun[2][8];
  v8f o[2][4] = {};
#pragma unroll
  for (int mt = 0; mt < 2; ++mt)
#pragma unroll
    for (int r = 0; r < 8; ++r) { mrun[mt][r] = -__builtin_inff(); lrun[mt][r] = 0.f; }

  for (int kb = 0; kb < SEQ; kb += 64) {
    // ---- S = Q K^T : 2(mt) x 4(nt) tiles, chained over two 32-wide d-chunks
    v8f s[2][4];
#pragma unroll
    for (int nt = 0; nt < 4; ++nt) {
      const h16_t* kp = K + (size_t)(b * SEQ + kb + nt * 16 + lr) * DIM + h * DHEAD;
      v16h bk0 = load_fragB(kp + hiL * 16);
      v16h bk1 = load_fragB(kp + 32 + hiL * 16);
#pragma unroll
      for (int mt = 0; mt < 2; ++mt) {
        v8f z = {};
        z = wmma_f16(aq[mt][0], bk0, z);
        s[mt][nt] = wmma_f16(aq[mt][1], bk1, z);
      }
    }

    // ---- online softmax: row max / exp / row sum (rows span 16 lanes)
    float mnew[2][8], alpha[2][8];
#pragma unroll
    for (int mt = 0; mt < 2; ++mt)
#pragma unroll
      for (int r = 0; r < 8; ++r) {
        float bm = fmaxf(fmaxf(s[mt][0][r], s[mt][1][r]),
                         fmaxf(s[mt][2][r], s[mt][3][r]));
#pragma unroll
        for (int off = 1; off < 16; off <<= 1)
          bm = fmaxf(bm, __shfl_xor(bm, off, 16));
        const float mn = fmaxf(mrun[mt][r], bm);
        mnew[mt][r]  = mn;
        alpha[mt][r] = __expf(mrun[mt][r] - mn);
      }
#pragma unroll
    for (int mt = 0; mt < 2; ++mt)
#pragma unroll
      for (int r = 0; r < 8; ++r) {
        const float mn = mnew[mt][r];
        float rs = 0.f;
#pragma unroll
        for (int nt = 0; nt < 4; ++nt) {
          const float p = __expf(s[mt][nt][r] - mn);
          s[mt][nt][r] = p;
          rs += p;
        }
#pragma unroll
        for (int off = 1; off < 16; off <<= 1)
          rs += __shfl_xor(rs, off, 16);
        lrun[mt][r] = lrun[mt][r] * alpha[mt][r] + rs;
        mrun[mt][r] = mn;
      }
#pragma unroll
    for (int mt = 0; mt < 2; ++mt)
#pragma unroll
      for (int nt = 0; nt < 4; ++nt)
#pragma unroll
        for (int r = 0; r < 8; ++r)
          o[mt][nt][r] *= alpha[mt][r];

    // ---- P: C-layout -> LDS -> A-fragment layout (f16)
#pragma unroll
    for (int mt = 0; mt < 2; ++mt)
#pragma unroll
      for (int nt = 0; nt < 4; ++nt)
#pragma unroll
        for (int r = 0; r < 8; ++r)
          P[(mt * 16 + r + hiL * 8) * 72 + nt * 16 + lr] = (h16_t)s[mt][nt][r];
    __syncthreads();
    v16h ap[2][2];
#pragma unroll
    for (int mt = 0; mt < 2; ++mt)
#pragma unroll
      for (int kc2 = 0; kc2 < 2; ++kc2)
        ap[mt][kc2] = load_fragA(P + (mt * 16 + lr) * 72 + kc2 * 32 + hiL * 16);

    // ---- O += P V  (V pre-transposed: contiguous over keys)
#pragma unroll
    for (int nt = 0; nt < 4; ++nt) {
      const h16_t* vp = Vt + ((size_t)(bh * DHEAD + nt * 16 + lr)) * SEQ + kb;
      v16h bv0 = load_fragB(vp + hiL * 16);
      v16h bv1 = load_fragB(vp + 32 + hiL * 16);
#pragma unroll
      for (int mt = 0; mt < 2; ++mt) {
        o[mt][nt] = wmma_f16(ap[mt][0], bv0, o[mt][nt]);
        o[mt][nt] = wmma_f16(ap[mt][1], bv1, o[mt][nt]);
      }
    }
    __syncthreads();
  }

  // ---- epilogue: O / l, store f32 (b, q, h*64+d)
#pragma unroll
  for (int mt = 0; mt < 2; ++mt) {
    float inv[8];
#pragma unroll
    for (int r = 0; r < 8; ++r) inv[r] = 1.0f / lrun[mt][r];
#pragma unroll
    for (int nt = 0; nt < 4; ++nt)
#pragma unroll
      for (int r = 0; r < 8; ++r) {
        const int q = q0 + mt * 16 + r + hiL * 8;
        const int d = nt * 16 + lr;
        Out[((size_t)(b * SEQ + q)) * DIM + h * DHEAD + d] = o[mt][nt][r] * inv[r];
      }
  }
}

// ---- host side --------------------------------------------------------------

extern "C" void kernel_launch(void* const* d_in, const int* in_sizes, int n_in,
                              void* d_out, int out_size, void* d_ws, size_t ws_size,
                              hipStream_t stream) {
  (void)in_sizes; (void)n_in; (void)out_size; (void)ws_size;
  const float* x   = (const float*)d_in[0];
  const float* lnw = (const float*)d_in[1];
  const float* lnb = (const float*)d_in[2];
  const float* wq  = (const float*)d_in[3];
  const float* bq  = (const float*)d_in[4];
  const float* wk  = (const float*)d_in[5];
  const float* bk  = (const float*)d_in[6];
  const float* wv  = (const float*)d_in[7];
  const float* bv  = (const float*)d_in[8];
  float* out = (float*)d_out;

  char* ws = (char*)d_ws;
  const size_t MB = (size_t)1 << 20;
  h16_t* h16  = (h16_t*)(ws);             // 16 MB: LN output f16 [8192,1024]
  h16_t* wtq  = (h16_t*)(ws + 16 * MB);   //  2 MB: wq^T f16
  h16_t* wtk  = (h16_t*)(ws + 18 * MB);   //  2 MB
  h16_t* wtv  = (h16_t*)(ws + 20 * MB);   //  2 MB
  h16_t* q16  = (h16_t*)(ws + 22 * MB);   // 16 MB: Q f16 [8192,1024]
  h16_t* k16  = (h16_t*)(ws + 38 * MB);   // 16 MB: K f16
  h16_t* vt16 = (h16_t*)(ws + 54 * MB);   // 16 MB: V f16 transposed [b,h,64,2048]

  ln_kernel<<<dim3(BATCH * SEQ), dim3(256), 0, stream>>>(x, lnw, lnb, h16);
  wt_kernel<<<dim3(4096), dim3(256), 0, stream>>>(wq, wtq);
  wt_kernel<<<dim3(4096), dim3(256), 0, stream>>>(wk, wtk);
  wt_kernel<<<dim3(4096), dim3(256), 0, stream>>>(wv, wtv);

  dim3 ggrid(64, 8);  // M tiles (8192/128) x N tiles (1024/128)
  gemm_qkv<0><<<ggrid, dim3(128), 0, stream>>>(h16, wtq, bq, q16);
  gemm_qkv<0><<<ggrid, dim3(128), 0, stream>>>(h16, wtk, bk, k16);
  gemm_qkv<1><<<ggrid, dim3(128), 0, stream>>>(h16, wtv, bv, vt16);

  attn_kernel<<<dim3(SEQ / 128, BATCH * HEADS), dim3(128), 0, stream>>>(
      q16, k16, vt16, out);
}
